// BlockTransformerPairBias_59115929862887
// MI455X (gfx1250) — matrix operand order repaired and checked by reading, not compiled, not run on hardware
//
#include <hip/hip_runtime.h>
#include <hip/hip_bf16.h>

// ---------------------------------------------------------------------------
// Problem constants (from reference setup_inputs)
// ---------------------------------------------------------------------------
// B=2, N=4096, Nres=1024, cs=512, cc=384, cz=128, H=8, block=64, nb=64, ch=64
#define RB 2
#define NN 4096
#define NRES 1024
#define CS 512
#define CC 384
#define CZ 128
#define NH 8
#define BLK 64
#define NB 64
#define CH 64
#define RTOT (RB * NN)      /* 8192 rows total */
#define RCOND (RB * NRES)   /* 2048 cond rows  */

typedef __attribute__((ext_vector_type(16))) __bf16 v16bf;
typedef __attribute__((ext_vector_type(8)))  float  v8f;

// ---------------------------------------------------------------------------
// Helpers
// ---------------------------------------------------------------------------
__device__ __forceinline__ unsigned short f2bf(float f) {
  unsigned int u = __builtin_bit_cast(unsigned int, f);
  u += 0x7FFFu + ((u >> 16) & 1u);   // round-to-nearest-even
  return (unsigned short)(u >> 16);
}
__device__ __forceinline__ __bf16 us2bf(unsigned short u) {
  return __builtin_bit_cast(__bf16, u);
}
__device__ __forceinline__ float wred_sum(float v) {
#pragma unroll
  for (int m = 16; m >= 1; m >>= 1) v += __shfl_xor(v, m, 32);
  return v;
}
__device__ __forceinline__ float wred_max(float v) {
#pragma unroll
  for (int m = 16; m >= 1; m >>= 1) v = fmaxf(v, __shfl_xor(v, m, 32));
  return v;
}
__device__ __forceinline__ float sigmoidf_(float x) {
  return 1.0f / (1.0f + __expf(-x));
}

__device__ __forceinline__ v8f wmma_bf16(v16bf a, v16bf b, v8f c) {
  return __builtin_amdgcn_wmma_f32_16x16x32_bf16(
      false, a, false, b, (short)0, c, false, false);
}

// --- CDNA5 async global->LDS copy (ASYNCcnt path, ISA 10 / 15.18.3) --------
__device__ __forceinline__ unsigned lds_off_of(const void* p) {
  // generic -> LDS(AS3) addrspacecast; AS3 pointers are 32-bit LDS offsets
  return (unsigned)(unsigned long long)
      (__attribute__((address_space(3))) const void*)p;
}
__device__ __forceinline__ void async_g2l_b128(unsigned lds_off,
                                               const void* gaddr) {
  asm volatile("global_load_async_to_lds_b128 %0, %1, off"
               :: "v"(lds_off), "v"((unsigned long long)(size_t)gaddr)
               : "memory");
}
__device__ __forceinline__ void wait_async0() {
  asm volatile("s_wait_asynccnt 0x0" ::: "memory");
}

// A fragment: 16x32 bf16, row-major source with row stride `ld` (ISA 7.12.2)
// lane<16: VGPR0..7 -> K=0..15 ; lane>=16: K=16..31 (contiguous order chosen here, B matches; not the ISA table order)
__device__ __forceinline__ v16bf load_a_frag(const unsigned short* A, int ld,
                                             int row, int kbase, int lane) {
  v16bf a;
  const int hi = (lane >= 16) ? 16 : 0;
  const unsigned short* p = A + row * ld + kbase;
#pragma unroll
  for (int j = 0; j < 8; ++j) {
    int kk = 2 * j + hi;
    a[2 * j]     = us2bf(p[kk]);
    a[2 * j + 1] = us2bf(p[kk + 1]);
  }
  return a;
}
// B fragment: 32x16 bf16, B stored row-major (K x N) with row stride `ld`
__device__ __forceinline__ v16bf load_b_frag(const unsigned short* B, int ld,
                                             int kbase, int col, int lane) {
  v16bf b;
  const int koff = (lane >= 16) ? 16 : 0;
  const unsigned short* p = B + (kbase + koff) * ld + col;
#pragma unroll
  for (int j = 0; j < 8; ++j) {
    b[2 * j]     = us2bf(p[(2 * j) * ld]);
    b[2 * j + 1] = us2bf(p[(2 * j + 1) * ld]);
  }
  return b;
}
// B fragment from transposed storage: element B(kk, col) = M[col*ld + kk]
__device__ __forceinline__ v16bf load_bT_frag(const unsigned short* M, int ld,
                                              int kbase, int col, int lane) {
  v16bf b;
  const int koff = (lane >= 16) ? 16 : 0;
  const unsigned short* p = M + col * ld + kbase + koff;
#pragma unroll
  for (int j = 0; j < 8; ++j) {
    b[2 * j]     = us2bf(p[2 * j]);
    b[2 * j + 1] = us2bf(p[2 * j + 1]);
  }
  return b;
}

// ---------------------------------------------------------------------------
// Weight packing / conversion kernels (tiny, run once per launch)
// ---------------------------------------------------------------------------
__global__ void pack_qkvg_kernel(const float* __restrict__ Wq,
                                 const float* __restrict__ Wkv,
                                 const float* __restrict__ Wg,
                                 unsigned short* __restrict__ out) {
  long i = (long)blockIdx.x * 256 + threadIdx.x;  // 512*2048
  if (i >= (long)CS * 2048) return;
  int k = (int)(i >> 11), n = (int)(i & 2047);
  float v = (n < 512) ? Wq[k * 512 + n]
          : (n < 1536) ? Wkv[k * 1024 + (n - 512)]
                       : Wg[k * 512 + (n - 1536)];
  out[i] = f2bf(v);
}
__global__ void pack_ada_kernel(const float* __restrict__ Wg,
                                const float* __restrict__ Wbs,
                                const float* __restrict__ Wt,
                                unsigned short* __restrict__ out) {
  long i = (long)blockIdx.x * 256 + threadIdx.x;  // 384*1536
  if (i >= (long)CC * 1536) return;
  int k = (int)(i / 1536), n = (int)(i % 1536);
  float v = (n < 512) ? Wg[k * 512 + n]
          : (n < 1024) ? Wbs[k * 512 + (n - 512)]
                       : Wt[k * 512 + (n - 1024)];
  out[i] = f2bf(v);
}
__global__ void pack_w12_kernel(const float* __restrict__ W1,
                                const float* __restrict__ W2,
                                unsigned short* __restrict__ out) {
  long i = (long)blockIdx.x * 256 + threadIdx.x;  // 512*2048
  if (i >= (long)CS * 2048) return;
  int k = (int)(i >> 11), n = (int)(i & 2047);
  float v = (n < 1024) ? W1[k * 1024 + n] : W2[k * 1024 + (n - 1024)];
  out[i] = f2bf(v);
}
__global__ void f32_to_bf16_kernel(const float* __restrict__ x,
                                   unsigned short* __restrict__ y, long n) {
  long i = (long)blockIdx.x * 256 + threadIdx.x;
  if (i < n) y[i] = f2bf(x[i]);
}

// ---------------------------------------------------------------------------
// Row LayerNorm -> bf16 (one wave per row, shuffle reductions)
// ---------------------------------------------------------------------------
__global__ void ln_rows_kernel(const float* __restrict__ X,
                               const float* __restrict__ w,
                               const float* __restrict__ b,
                               unsigned short* __restrict__ Y, int L) {
  int row  = blockIdx.x * 8 + (threadIdx.x >> 5);
  int lane = threadIdx.x & 31;
  const float* x = X + (long)row * L;
  float s = 0.f, s2 = 0.f;
  for (int c = lane; c < L; c += 32) { float v = x[c]; s += v; s2 += v * v; }
  s = wred_sum(s); s2 = wred_sum(s2);
  float mean = s / L;
  float var  = s2 / L - mean * mean;
  float inv  = rsqrtf(var + 1e-5f);
  unsigned short* y = Y + (long)row * L;
  for (int c = lane; c < L; c += 32) {
    float v = (x[c] - mean) * inv;
    if (w) v *= w[c];
    if (b) v += b[c];
    y[c] = f2bf(v);
  }
}

// ---------------------------------------------------------------------------
// Generic bf16 WMMA GEMM: C(f32, MxN) = A(bf16, MxK) @ B(bf16, KxN)
// 64x64 workgroup tile, 8 waves x two 16x16 tiles, K-step 32.
// Tiles staged with async global->LDS B128 DMA + next-tile prefetch.
// ---------------------------------------------------------------------------
__global__ __launch_bounds__(256)
void gemm_bf16_kernel(const unsigned short* __restrict__ A,
                      const unsigned short* __restrict__ Bw,
                      float* __restrict__ C, int M, int N, int K) {
  __shared__ __align__(16) unsigned short As[64 * 32];
  __shared__ __align__(16) unsigned short Bs[32 * 64];
  const int tid  = threadIdx.x;
  const int lane = tid & 31;
  const int wave = tid >> 5;
  const int m0 = blockIdx.y * 64;
  const int n0 = blockIdx.x * 64;
  const int t0 = wave, t1 = wave + 8;
  const int ti0 = t0 >> 2, tj0 = t0 & 3;
  const int ti1 = t1 >> 2, tj1 = t1 & 3;
  // per-thread staging coordinates (each thread moves 16 contiguous bytes)
  const int ar = tid >> 2, ac = (tid & 3) * 8;
  const int br = tid >> 3, bc = (tid & 7) * 8;
  const unsigned a_lds = lds_off_of(As + ar * 32 + ac);
  const unsigned b_lds = lds_off_of(Bs + br * 64 + bc);
  v8f acc0 = {}; v8f acc1 = {};
  for (int kb = 0; kb < K; kb += 32) {
    __syncthreads();
    // async DMA: global -> LDS, 16B per lane, tracked by ASYNCcnt
    async_g2l_b128(a_lds, A + (long)(m0 + ar) * K + kb + ac);
    async_g2l_b128(b_lds, Bw + (long)(kb + br) * N + n0 + bc);
    if (kb + 32 < K) {  // pull next K-tile toward L2 while WMMA runs
      __builtin_prefetch(A + (long)(m0 + ar) * K + kb + 32 + ac, 0, 1);
      __builtin_prefetch(Bw + (long)(kb + 32 + br) * N + n0 + bc, 0, 1);
    }
    wait_async0();
    __syncthreads();
    v16bf a0 = load_a_frag(As, 32, ti0 * 16 + (lane & 15), 0, lane);
    v16bf b0 = load_b_frag(Bs, 64, 0, tj0 * 16 + (lane & 15), lane);
    acc0 = wmma_bf16(a0, b0, acc0);
    v16bf a1 = load_a_frag(As, 32, ti1 * 16 + (lane & 15), 0, lane);
    v16bf b1 = load_b_frag(Bs, 64, 0, tj1 * 16 + (lane & 15), lane);
    acc1 = wmma_bf16(a1, b1, acc1);
  }
  const int hi = (lane >= 16) ? 8 : 0;
#pragma unroll
  for (int r = 0; r < 8; ++r) {
    C[(long)(m0 + ti0 * 16 + r + hi) * N + n0 + tj0 * 16 + (lane & 15)] = acc0[r];
    C[(long)(m0 + ti1 * 16 + r + hi) * N + n0 + tj1 * 16 + (lane & 15)] = acc1[r];
  }
}

// ---------------------------------------------------------------------------
// Pair bias: LN(framepair, cz=128) @ Wbias(128x8) -> (B*nb*H, 64, 64)
// one wave per (b,g,i,j) row
// ---------------------------------------------------------------------------
__global__ void bias_kernel(const float* __restrict__ fp,
                            const float* __restrict__ w,
                            const float* __restrict__ bv,
                            const float* __restrict__ Wbias,
                            float* __restrict__ out) {
  int row  = blockIdx.x * 8 + (threadIdx.x >> 5);   // 0 .. B*nb*64*64-1
  int lane = threadIdx.x & 31;
  const float* x = fp + (long)row * CZ;
  float v[4], s = 0.f, s2 = 0.f;
#pragma unroll
  for (int t = 0; t < 4; ++t) { v[t] = x[lane + 32 * t]; s += v[t]; s2 += v[t] * v[t]; }
  s = wred_sum(s); s2 = wred_sum(s2);
  float mean = s * (1.0f / CZ);
  float var  = s2 * (1.0f / CZ) - mean * mean;
  float inv  = rsqrtf(var + 1e-5f);
  float y[4];
#pragma unroll
  for (int t = 0; t < 4; ++t) {
    int c = lane + 32 * t;
    y[t] = (v[t] - mean) * inv * w[c] + bv[c];
  }
  int bg = row >> 12, i = (row >> 6) & 63, j = row & 63;
#pragma unroll
  for (int h = 0; h < NH; ++h) {
    float p = 0.f;
#pragma unroll
    for (int t = 0; t < 4; ++t) p += y[t] * Wbias[(lane + 32 * t) * NH + h];
    p = wred_sum(p);
    if (lane == 0)
      out[((long)(bg * NH + h)) * 4096 + i * 64 + j] = p;
  }
}

// ---------------------------------------------------------------------------
// Fused per-(block,head) attention: scores(WMMA) + bias + mask + softmax +
// P@V(WMMA) + sigmoid-gate epilogue. grid.x = B*nb*H = 1024, block = 256.
// ---------------------------------------------------------------------------
__global__ __launch_bounds__(256)
void attn_kernel(const float* __restrict__ qkvg,   // (8192, 2048) f32
                 const float* __restrict__ biasB,  // (B*nb*H, 64, 64) f32
                 const float* __restrict__ mask,   // (B*N) f32
                 unsigned short* __restrict__ obf) // (8192, 512) bf16 (gated)
{
  __shared__ __align__(16) unsigned short qs[64 * 64];
  __shared__ __align__(16) unsigned short ks[64 * 64];
  __shared__ __align__(16) unsigned short vs[64 * 64];
  __shared__ float sc[64 * 64];
  __shared__ __align__(16) unsigned short pb[64 * 64];
  __shared__ float mrow[64];

  const int bgh = blockIdx.x;
  const int h   = bgh & 7;
  const int bg  = bgh >> 3;
  const int tid = threadIdx.x, lane = tid & 31, wave = tid >> 5;
  const long rowbase = (long)bg * 64;

  for (int e = tid; e < 64 * 64; e += 256) {
    int i = e >> 6, c = e & 63;
    const float* rp = qkvg + (rowbase + i) * 2048;
    qs[e] = f2bf(rp[h * 64 + c]);
    ks[e] = f2bf(rp[512 + h * 64 + c]);
    vs[e] = f2bf(rp[1024 + h * 64 + c]);
  }
  if (tid < 64) mrow[tid] = mask[rowbase + tid];
  __syncthreads();

  const int t0 = wave, t1 = wave + 8;
  const int ti0 = t0 >> 2, tj0 = t0 & 3;
  const int ti1 = t1 >> 2, tj1 = t1 & 3;
  const int hi = (lane >= 16) ? 8 : 0;

  // scores = q @ k^T
  v8f a0 = {}; v8f a1 = {};
#pragma unroll
  for (int kb = 0; kb < 64; kb += 32) {
    v16bf af = load_a_frag(qs, 64, ti0 * 16 + (lane & 15), kb, lane);
    v16bf bf = load_bT_frag(ks, 64, kb, tj0 * 16 + (lane & 15), lane);
    a0 = wmma_bf16(af, bf, a0);
    af = load_a_frag(qs, 64, ti1 * 16 + (lane & 15), kb, lane);
    bf = load_bT_frag(ks, 64, kb, tj1 * 16 + (lane & 15), lane);
    a1 = wmma_bf16(af, bf, a1);
  }
  const float* bp = biasB + (long)bgh * 4096;
#pragma unroll
  for (int r = 0; r < 8; ++r) {
    int row = ti0 * 16 + r + hi, col = tj0 * 16 + (lane & 15);
    sc[row * 64 + col] = a0[r] * 0.125f + bp[row * 64 + col]
                       + (mrow[row] * mrow[col] - 1.0f) * 100000.0f;
    row = ti1 * 16 + r + hi; col = tj1 * 16 + (lane & 15);
    sc[row * 64 + col] = a1[r] * 0.125f + bp[row * 64 + col]
                       + (mrow[row] * mrow[col] - 1.0f) * 100000.0f;
  }
  __syncthreads();

  // softmax over j (each wave owns 8 rows; 2 cols per lane)
#pragma unroll
  for (int rr = 0; rr < 8; ++rr) {
    int row = wave * 8 + rr;
    float x0 = sc[row * 64 + lane], x1 = sc[row * 64 + 32 + lane];
    float mx = wred_max(fmaxf(x0, x1));
    float e0 = __expf(x0 - mx), e1 = __expf(x1 - mx);
    float sum = wred_sum(e0 + e1);
    float invs = 1.0f / sum;
    pb[row * 64 + lane]      = f2bf(e0 * invs);
    pb[row * 64 + 32 + lane] = f2bf(e1 * invs);
  }
  __syncthreads();

  // o = p @ v, then gate with sigmoid(x @ Wgate)
  v8f o0 = {}; v8f o1 = {};
#pragma unroll
  for (int kb = 0; kb < 64; kb += 32) {
    v16bf af = load_a_frag(pb, 64, ti0 * 16 + (lane & 15), kb, lane);
    v16bf bf = load_b_frag(vs, 64, kb, tj0 * 16 + (lane & 15), lane);
    o0 = wmma_bf16(af, bf, o0);
    af = load_a_frag(pb, 64, ti1 * 16 + (lane & 15), kb, lane);
    bf = load_b_frag(vs, 64, kb, tj1 * 16 + (lane & 15), lane);
    o1 = wmma_bf16(af, bf, o1);
  }
#pragma unroll
  for (int r = 0; r < 8; ++r) {
    {
      int row = ti0 * 16 + r + hi, col = tj0 * 16 + (lane & 15);
      long grow = rowbase + row;
      float g = qkvg[grow * 2048 + 1536 + h * 64 + col];
      obf[grow * 512 + h * 64 + col] = f2bf(o0[r] * sigmoidf_(g));
    }
    {
      int row = ti1 * 16 + r + hi, col = tj1 * 16 + (lane & 15);
      long grow = rowbase + row;
      float g = qkvg[grow * 2048 + 1536 + h * 64 + col];
      obf[grow * 512 + h * 64 + col] = f2bf(o1[r] * sigmoidf_(g));
    }
  }
}

// ---------------------------------------------------------------------------
// Pointwise epilogues
// ---------------------------------------------------------------------------
__global__ void h_kernel(const float* __restrict__ re,
                         const float* __restrict__ upd,
                         const float* __restrict__ mask,
                         float* __restrict__ hbuf) {
  long i = (long)blockIdx.x * 256 + threadIdx.x;   // RTOT*CS
  int row = (int)(i >> 9);
  hbuf[i] = re[i] + upd[i] * mask[row];
}

// t = LN(h) * sigmoid((condp_gate[idx]+b)*m) + condp_bias[idx]*m  (wave/row)
__global__ void t_kernel(const float* __restrict__ hbuf,
                         const float* __restrict__ condp,
                         const int* __restrict__ ridx,
                         const float* __restrict__ mask,
                         const float* __restrict__ b_ada_gate,
                         unsigned short* __restrict__ tbf) {
  int row  = blockIdx.x * 8 + (threadIdx.x >> 5);  // 0..8191
  int lane = threadIdx.x & 31;
  int b    = row >> 12;
  int id   = ridx[row];
  float m  = mask[row];
  const float* cp = condp + ((long)(b * NRES + id)) * 1536;
  const float* x  = hbuf + (long)row * CS;
  float s = 0.f, s2 = 0.f;
  for (int c = lane; c < CS; c += 32) { float v = x[c]; s += v; s2 += v * v; }
  s = wred_sum(s); s2 = wred_sum(s2);
  float mean = s * (1.0f / CS);
  float inv  = rsqrtf(s2 * (1.0f / CS) - mean * mean + 1e-5f);
  unsigned short* y = tbf + (long)row * CS;
  for (int c = lane; c < CS; c += 32) {
    float g    = (cp[c] + b_ada_gate[c]) * m;
    float bias = cp[512 + c] * m;
    float v    = (x[c] - mean) * inv;
    y[c] = f2bf(v * sigmoidf_(g) + bias);
  }
}

__global__ void bb_kernel(const float* __restrict__ u,
                          unsigned short* __restrict__ bb) {
  long i = (long)blockIdx.x * 256 + threadIdx.x;   // RTOT*1024
  int r = (int)(i >> 10), c = (int)(i & 1023);
  float u1 = u[(long)r * 2048 + c];
  float u2 = u[(long)r * 2048 + 1024 + c];
  bb[i] = f2bf(u1 * sigmoidf_(u1) * u2);           // silu(u1)*u2
}

__global__ void final_kernel(const float* __restrict__ hbuf,
                             const float* __restrict__ trans,
                             const float* __restrict__ condp,
                             const int* __restrict__ ridx,
                             const float* __restrict__ mask,
                             const float* __restrict__ b_tgate,
                             float* __restrict__ out) {
  long i = (long)blockIdx.x * 256 + threadIdx.x;   // RTOT*CS
  int row = (int)(i >> 9), c = (int)(i & 511);
  int b = row >> 12;
  int id = ridx[row];
  float m = mask[row];
  float g = (condp[((long)(b * NRES + id)) * 1536 + 1024 + c] + b_tgate[c]) * m;
  out[i] = hbuf[i] + sigmoidf_(g) * trans[i] * m;
}

// ---------------------------------------------------------------------------
// Workspace layout (with aliasing: QKVG->U, BIAS->TRANS, XBF->TBF, UPD->BB)
// ---------------------------------------------------------------------------
constexpr size_t OFF_WQKVG = 0;
constexpr size_t SZ_WQKVG  = 512ull * 2048 * 2;
constexpr size_t OFF_WOUT  = OFF_WQKVG + SZ_WQKVG;
constexpr size_t SZ_WOUT   = 512ull * 512 * 2;
constexpr size_t OFF_WADA  = OFF_WOUT + SZ_WOUT;
constexpr size_t SZ_WADA   = 384ull * 1536 * 2;
constexpr size_t OFF_W12   = OFF_WADA + SZ_WADA;
constexpr size_t SZ_W12    = 512ull * 2048 * 2;
constexpr size_t OFF_WB    = OFF_W12 + SZ_W12;
constexpr size_t SZ_WB     = 1024ull * 512 * 2;
constexpr size_t OFF_XBF   = OFF_WB + SZ_WB;          // reused as TBF
constexpr size_t SZ_XBF    = (size_t)RTOT * CS * 2;
constexpr size_t OFF_QKVG  = OFF_XBF + SZ_XBF;        // reused as U
constexpr size_t SZ_QKVG   = (size_t)RTOT * 2048 * 4;
constexpr size_t OFF_BIAS  = OFF_QKVG + SZ_QKVG;      // reused as TRANS
constexpr size_t SZ_BIAS   = 1024ull * 4096 * 4;
constexpr size_t OFF_OBF   = OFF_BIAS + SZ_BIAS;
constexpr size_t SZ_OBF    = (size_t)RTOT * CS * 2;
constexpr size_t OFF_UPD   = OFF_OBF + SZ_OBF;        // reused as BB
constexpr size_t SZ_UPD    = (size_t)RTOT * CS * 4;
constexpr size_t OFF_HF    = OFF_UPD + SZ_UPD;
constexpr size_t SZ_HF     = (size_t)RTOT * CS * 4;
constexpr size_t OFF_CBF   = OFF_HF + SZ_HF;
constexpr size_t SZ_CBF    = (size_t)RCOND * CC * 2;
constexpr size_t OFF_CONDP = OFF_CBF + SZ_CBF;

extern "C" void kernel_launch(void* const* d_in, const int* in_sizes, int n_in,
                              void* d_out, int out_size, void* d_ws, size_t ws_size,
                              hipStream_t stream) {
  const float* s_in       = (const float*)d_in[0];
  const float* rigids     = (const float*)d_in[1];
  const float* framepair  = (const float*)d_in[2];
  const float* mask       = (const float*)d_in[3];
  const int*   ridx       = (const int*)d_in[4];
  const float* ln_w       = (const float*)d_in[6];
  const float* ln_b       = (const float*)d_in[7];
  const float* Wq         = (const float*)d_in[8];
  const float* Wkv        = (const float*)d_in[9];
  const float* bias_ln_w  = (const float*)d_in[10];
  const float* bias_ln_b  = (const float*)d_in[11];
  const float* Wbias      = (const float*)d_in[12];
  const float* Wgate      = (const float*)d_in[13];
  const float* Wout       = (const float*)d_in[14];
  const float* ada_w      = (const float*)d_in[15];
  const float* W_ada_gate = (const float*)d_in[16];
  const float* b_ada_gate = (const float*)d_in[17];
  const float* W_ada_bias = (const float*)d_in[18];
  const float* W1         = (const float*)d_in[19];
  const float* W2         = (const float*)d_in[20];
  const float* W_tgate    = (const float*)d_in[21];
  const float* b_tgate    = (const float*)d_in[22];
  const float* Wb         = (const float*)d_in[23];
  float* out = (float*)d_out;

  char* ws = (char*)d_ws;
  unsigned short* wqkvg = (unsigned short*)(ws + OFF_WQKVG);
  unsigned short* wout  = (unsigned short*)(ws + OFF_WOUT);
  unsigned short* wada  = (unsigned short*)(ws + OFF_WADA);
  unsigned short* w12   = (unsigned short*)(ws + OFF_W12);
  unsigned short* wb    = (unsigned short*)(ws + OFF_WB);
  unsigned short* xbf   = (unsigned short*)(ws + OFF_XBF);
  unsigned short* tbf   = xbf;                               // alias
  float*          qkvg  = (float*)(ws + OFF_QKVG);
  float*          ubuf  = qkvg;                              // alias
  float*          biasb = (float*)(ws + OFF_BIAS);
  float*          trans = biasb;                             // alias
  unsigned short* obf   = (unsigned short*)(ws + OFF_OBF);
  float*          updf  = (float*)(ws + OFF_UPD);
  unsigned short* bbuf  = (unsigned short*)(ws + OFF_UPD);   // alias
  float*          hbuf  = (float*)(ws + OFF_HF);
  unsigned short* cbf   = (unsigned short*)(ws + OFF_CBF);
  float*          condp = (float*)(ws + OFF_CONDP);

  // --- weight conversion / packing ---
  pack_qkvg_kernel<<<(512 * 2048 + 255) / 256, 256, 0, stream>>>(Wq, Wkv, Wgate, wqkvg);
  f32_to_bf16_kernel<<<(512 * 512 + 255) / 256, 256, 0, stream>>>(Wout, wout, 512l * 512);
  pack_ada_kernel<<<(384 * 1536 + 255) / 256, 256, 0, stream>>>(W_ada_gate, W_ada_bias, W_tgate, wada);
  pack_w12_kernel<<<(512 * 2048 + 255) / 256, 256, 0, stream>>>(W1, W2, w12);
  f32_to_bf16_kernel<<<(1024 * 512 + 255) / 256, 256, 0, stream>>>(Wb, wb, 1024l * 512);

  // --- x = LN(rigids_embed) -> bf16 ---
  ln_rows_kernel<<<RTOT / 8, 256, 0, stream>>>(rigids, ln_w, ln_b, xbf, CS);

  // --- qkvg = x @ [Wq|Wkv|Wgate]  (8192 x 2048, K=512) ---
  gemm_bf16_kernel<<<dim3(2048 / 64, RTOT / 64), 256, 0, stream>>>(
      xbf, wqkvg, qkvg, RTOT, 2048, CS);

  // --- pair bias (streams the 268 MB tensor once) ---
  bias_kernel<<<(RB * NB * BLK * BLK) / 8, 256, 0, stream>>>(
      framepair, bias_ln_w, bias_ln_b, Wbias, biasb);

  // --- fused block attention + gate ---
  attn_kernel<<<RB * NB * NH, 256, 0, stream>>>(qkvg, biasb, mask, obf);

  // --- update = o_gated @ Wout ; h = rigids + update*mask ---
  gemm_bf16_kernel<<<dim3(CS / 64, RTOT / 64), 256, 0, stream>>>(
      obf, wout, updf, RTOT, CS, CS);
  h_kernel<<<(RTOT * CS) / 256, 256, 0, stream>>>(rigids, updf, mask, hbuf);

  // --- cond = LN(s, ada_w) ; condp = cond @ [Wg|Wbias|Wtg] ---
  ln_rows_kernel<<<RCOND / 8, 256, 0, stream>>>(s_in, ada_w, nullptr, cbf, CC);
  gemm_bf16_kernel<<<dim3(1536 / 64, RCOND / 64), 256, 0, stream>>>(
      cbf, wada, condp, RCOND, 1536, CC);

  // --- t = AdaLN(h) -> bf16 ---
  t_kernel<<<RTOT / 8, 256, 0, stream>>>(hbuf, condp, ridx, mask, b_ada_gate, tbf);

  // --- u = t @ [W1|W2] ; bb = silu(u1)*u2 ---
  gemm_bf16_kernel<<<dim3(2048 / 64, RTOT / 64), 256, 0, stream>>>(
      tbf, w12, ubuf, RTOT, 2048, CS);
  bb_kernel<<<(RTOT * 1024) / 256, 256, 0, stream>>>(ubuf, bbuf);

  // --- trans_raw = bb @ Wb ; out = h + sigmoid(tgate)*trans*mask ---
  gemm_bf16_kernel<<<dim3(CS / 64, RTOT / 64), 256, 0, stream>>>(
      bbuf, wb, trans, RTOT, CS, 1024);
  final_kernel<<<(RTOT * CS) / 256, 256, 0, stream>>>(
      hbuf, trans, condp, ridx, mask, b_tgate, out);
}
